// MultiHeadSelfAttention_37744172597633
// MI455X (gfx1250) — matrix operand order repaired and checked by reading, not compiled, not run on hardware
//
#include <hip/hip_runtime.h>

// ---------------------------------------------------------------------------
// MHSA for MI455X (gfx1250, wave32) built on V_WMMA_F32_16X16X32_BF16.
// Pipeline: cvt(fp32->bf16) -> 3x proj GEMM -> column-softmax stats ->
//           ctx (scores+softmax+PV) -> output proj GEMM.
// ---------------------------------------------------------------------------

typedef __bf16 bf16;
typedef __attribute__((ext_vector_type(16))) bf16  v16bf;
typedef __attribute__((ext_vector_type(8)))  bf16  v8bf;
typedef __attribute__((ext_vector_type(8)))  float v8f;

#define B_  2
#define S_  2048
#define D_  1024
#define H_  16
#define HD_ 64
#define BH_ (B_ * H_)
#define NEGV (-1000000000.0f)
#define SCL  (0.125f)   // 1/sqrt(64)

// ---- fragment helpers ------------------------------------------------------
__device__ __forceinline__ v16bf cat8(v8bf lo, v8bf hi) {
  return __builtin_shufflevector(lo, hi, 0,1,2,3,4,5,6,7,8,9,10,11,12,13,14,15);
}
// A fragment (16x32, M=lane&15): K runs {ka..ka+7} and {16+ka..16+ka+7}
__device__ __forceinline__ v16bf afrag_bf(const bf16* p, int ka) {
  v8bf lo = *(const v8bf*)(p + ka);
  v8bf hi = *(const v8bf*)(p + 16 + ka);
  return cat8(lo, hi);
}
// B fragment (32x16, N=lane&15): K runs {kb/2..kb/2+7} and {16+kb/2..16+kb/2+7}
__device__ __forceinline__ v16bf bfrag_bf(const bf16* p, int kb) {
  v8bf lo = *(const v8bf*)(p + (kb >> 1));
  v8bf hi = *(const v8bf*)(p + (kb >> 1) + 16);
  return cat8(lo, hi);
}
__device__ __forceinline__ v8f wmma_bf16(v16bf a, v16bf b, v8f c) {
  return __builtin_amdgcn_wmma_f32_16x16x32_bf16(false, a, false, b,
                                                 (short)0, c, false, false);
}

// ---------------------------------------------------------------------------
// Kernel 0: fp32 -> bf16 elementwise (8 elems/thread, b128 in/out). Pure
// memory-bound; pays conversion cost once instead of per GEMM-tile reuse.
// ---------------------------------------------------------------------------
__global__ __launch_bounds__(256) void k_cvt(const float* __restrict__ in,
                                             bf16* __restrict__ out, int n8) {
  const int i = blockIdx.x * 256 + threadIdx.x;
  if (i >= n8) return;
  const float4* p = (const float4*)in + (size_t)i * 2;
  const float4 a = p[0], b = p[1];
  v8bf r;
  r[0] = (bf16)a.x; r[1] = (bf16)a.y; r[2] = (bf16)a.z; r[3] = (bf16)a.w;
  r[4] = (bf16)b.x; r[5] = (bf16)b.y; r[6] = (bf16)b.z; r[7] = (bf16)b.w;
  *(v8bf*)(out + (size_t)i * 8) = r;
}

// ---------------------------------------------------------------------------
// Kernel 1: y = x @ W^T + b  (bf16 in, bf16 out).  One wave = 16(M) x 64(N)
// tile via 4 round-robin accumulators (hides WMMA hazard, 4x A reuse).
// vmode==0: store [b,h,s,hd].  vmode==1: store [b,h,hd,s] (vector store).
// Each 64-wide N group is exactly one head (HD==64).
// ---------------------------------------------------------------------------
__global__ __launch_bounds__(128) void k_proj(const bf16* __restrict__ xb,
                                              const bf16* __restrict__ Wb,
                                              const float* __restrict__ bias,
                                              bf16* __restrict__ out, int vmode) {
  const int lane = threadIdx.x;
  const int r  = lane & 15;
  const int ka = (lane >> 4) * 8;
  const int kb = (lane >> 4) * 16;
  const int eg    = blockIdx.x;                     // 0..15  (64-wide N group = head)
  const int mtile = blockIdx.y * 4 + threadIdx.y;   // 0..255

  const bf16* xrow = xb + (size_t)(mtile * 16 + r) * D_;
  const bf16* wrow[4];
#pragma unroll
  for (int et = 0; et < 4; ++et)
    wrow[et] = Wb + (size_t)(eg * 64 + et * 16 + r) * D_;

  v8f acc[4] = {};
  for (int kc = 0; kc < D_; kc += 32) {
    const v16bf va = afrag_bf(xrow + kc, ka);
#pragma unroll
    for (int et = 0; et < 4; ++et)
      acc[et] = wmma_bf16(va, bfrag_bf(wrow[et] + kc, kb), acc[et]);
  }

  const int m0 = mtile * 16 + ka;                   // first of 8 rows this lane holds
  const int h  = eg;
#pragma unroll
  for (int et = 0; et < 4; ++et) {
    const int ecol = eg * 64 + et * 16 + r;
    const float bv = bias[ecol];
    const int dh = et * 16 + r;
    if (vmode) {                                    // [b,h,hd,s]
      const int b = m0 >> 11, s = m0 & (S_ - 1);
      v8bf pk;
#pragma unroll
      for (int i = 0; i < 8; ++i) pk[i] = (bf16)(acc[et][i] + bv);
      *(v8bf*)(out + ((size_t)(b * H_ + h) * HD_ + dh) * S_ + s) = pk;
    } else {                                        // [b,h,s,hd]
#pragma unroll
      for (int i = 0; i < 8; ++i) {
        const int m = m0 + i, b = m >> 11, s = m & (S_ - 1);
        out[((size_t)(b * H_ + h) * S_ + s) * HD_ + dh] = (bf16)(acc[et][i] + bv);
      }
    }
  }
}

// ---------------------------------------------------------------------------
// Kernel 2: column softmax stats.  scores[q,k] = (Qh.Kh^T)/8, mask k>=q -> NEG,
// softmax over q (axis=-2).  One wave owns 16 key columns, streams all
// q-tiles with online max/sum; lane halves of a column merge via shfl_xor 16.
// ---------------------------------------------------------------------------
__global__ __launch_bounds__(128) void k_stats(const bf16* __restrict__ Qh,
                                               const bf16* __restrict__ Kh,
                                               float* __restrict__ mcol,
                                               float* __restrict__ zcol) {
  const int lane = threadIdx.x;
  const int r  = lane & 15;
  const int ka = (lane >> 4) * 8;
  const int kb = (lane >> 4) * 16;
  const int ktile = blockIdx.x * 4 + threadIdx.y;   // 0..127
  const int bh = blockIdx.y;                        // 0..31

  const int kglob = ktile * 16 + r;                 // key column owned
  const bf16* qbase = Qh + (size_t)bh * S_ * HD_;
  const bf16* krow  = Kh + ((size_t)bh * S_ + kglob) * HD_;
  const v16bf kb0 = bfrag_bf(krow, kb);
  const v16bf kb1 = bfrag_bf(krow + 32, kb);

  float runm = -__builtin_inff(), runz = 0.0f;

  for (int qt = 0; qt < S_ / 16; ++qt) {
    const bf16* qrow = qbase + (size_t)(qt * 16 + r) * HD_;
    v8f sc = {};
    sc = wmma_bf16(afrag_bf(qrow, ka), kb0, sc);
    sc = wmma_bf16(afrag_bf(qrow + 32, ka), kb1, sc);

    const int q0 = qt * 16 + ka;                    // rows held by this lane
    float sv[8], tmax = -__builtin_inff();
#pragma unroll
    for (int i = 0; i < 8; ++i) {
      const float s = (kglob >= q0 + i) ? NEGV : sc[i] * SCL;
      sv[i] = s;
      tmax = fmaxf(tmax, s);
    }
    tmax = fmaxf(tmax, __shfl_xor(tmax, 16, 32));   // merge column halves
    const float newm = fmaxf(runm, tmax);
    float add = 0.0f;
#pragma unroll
    for (int i = 0; i < 8; ++i) add += __expf(sv[i] - newm);
    runz = runz * __expf(runm - newm) + add;
    runm = newm;
  }
  runz += __shfl_xor(runz, 16, 32);
  if (lane < 16) {
    mcol[(size_t)bh * S_ + kglob] = runm;
    zcol[(size_t)bh * S_ + kglob] = runz;
  }
}

// ---------------------------------------------------------------------------
// Kernel 3: ctx[q,:] = sum_k p[q,k] * V[k,:]  with p = exp(s-m[k])/Z[k].
// One wave -> 16 q-rows x full head dim (64).  Per 32-key chunk: 4 WMMAs for
// scores, probabilities bounced through per-wave LDS (D-layout -> A-layout),
// then 4 independent WMMAs against V^T.  ctx stored bf16 in [b,s,D].
// ---------------------------------------------------------------------------
__global__ __launch_bounds__(128) void k_ctx(const bf16* __restrict__ Qh,
                                             const bf16* __restrict__ Kh,
                                             const bf16* __restrict__ VhT,
                                             const float* __restrict__ mcol,
                                             const float* __restrict__ zcol,
                                             bf16* __restrict__ ctx) {
  __shared__ __align__(16) bf16 sP[4][16 * 32];     // per-wave p tile
  const int lane = threadIdx.x;
  const int r  = lane & 15;
  const int ka = (lane >> 4) * 8;
  const int kb = (lane >> 4) * 16;
  const int qtile = blockIdx.x * 4 + threadIdx.y;   // 0..127
  const int bh = blockIdx.y;
  const int b = bh >> 4, h = bh & 15;

  bf16* myl = sP[threadIdx.y];
  const bf16* qrow = Qh + ((size_t)bh * S_ + qtile * 16 + r) * HD_;
  const v16bf qa0 = afrag_bf(qrow, ka);
  const v16bf qa1 = afrag_bf(qrow + 32, ka);

  v8f acc[4] = {};

  for (int kc = 0; kc < S_; kc += 32) {
#pragma unroll
    for (int sub = 0; sub < 2; ++sub) {
      const int key = kc + sub * 16 + r;            // score column owned
      const bf16* krow = Kh + ((size_t)bh * S_ + key) * HD_;
      v8f sc = {};
      sc = wmma_bf16(qa0, bfrag_bf(krow, kb), sc);
      sc = wmma_bf16(qa1, bfrag_bf(krow + 32, kb), sc);

      const float mc = mcol[(size_t)bh * S_ + key];
      const float iz = 1.0f / zcol[(size_t)bh * S_ + key];
      const int q0 = qtile * 16 + ka;
#pragma unroll
      for (int i = 0; i < 8; ++i) {
        const float s = (key >= q0 + i) ? NEGV : sc[i] * SCL;
        myl[(ka + i) * 32 + sub * 16 + r] = (bf16)(__expf(s - mc) * iz);
      }
    }
    // same-wave LDS store->load transpose; enforce DS ordering explicitly
    asm volatile("s_wait_dscnt 0" ::: "memory");
    const v16bf pa = afrag_bf(myl + r * 32, ka);
#pragma unroll
    for (int dt = 0; dt < 4; ++dt) {
      const bf16* vrow = VhT + ((size_t)bh * HD_ + dt * 16 + r) * S_ + kc;
      acc[dt] = wmma_bf16(pa, bfrag_bf(vrow, kb), acc[dt]);
    }
    asm volatile("" ::: "memory");
  }

  // store: row = b*S + q, col = h*64 + d
#pragma unroll
  for (int dt = 0; dt < 4; ++dt) {
    const int dcol = h * HD_ + dt * 16 + r;
#pragma unroll
    for (int i = 0; i < 8; ++i) {
      const int q = qtile * 16 + ka + i;
      ctx[(size_t)(b * S_ + q) * D_ + dcol] = (bf16)acc[dt][i];
    }
  }
}

// ---------------------------------------------------------------------------
// Kernel 4: out = ctx @ Wo^T + bo  (bf16 A/B, fp32 out), 16x64 per wave.
// ---------------------------------------------------------------------------
__global__ __launch_bounds__(128) void k_oproj(const bf16* __restrict__ ctx,
                                               const bf16* __restrict__ Wob,
                                               const float* __restrict__ bo,
                                               float* __restrict__ out) {
  const int lane = threadIdx.x;
  const int r  = lane & 15;
  const int ka = (lane >> 4) * 8;
  const int kb = (lane >> 4) * 16;
  const int eg    = blockIdx.x;
  const int mtile = blockIdx.y * 4 + threadIdx.y;

  const bf16* xrow = ctx + (size_t)(mtile * 16 + r) * D_;
  const bf16* wrow[4];
#pragma unroll
  for (int et = 0; et < 4; ++et)
    wrow[et] = Wob + (size_t)(eg * 64 + et * 16 + r) * D_;

  v8f acc[4] = {};
  for (int kc = 0; kc < D_; kc += 32) {
    const v16bf va = afrag_bf(xrow + kc, ka);
#pragma unroll
    for (int et = 0; et < 4; ++et)
      acc[et] = wmma_bf16(va, bfrag_bf(wrow[et] + kc, kb), acc[et]);
  }

  const int m0 = mtile * 16 + ka;
#pragma unroll
  for (int et = 0; et < 4; ++et) {
    const int ecol = eg * 64 + et * 16 + r;
    const float bv = bo[ecol];
#pragma unroll
    for (int i = 0; i < 8; ++i)
      out[(size_t)(m0 + i) * D_ + ecol] = acc[et][i] + bv;
  }
}

// ---------------------------------------------------------------------------
extern "C" void kernel_launch(void* const* d_in, const int* in_sizes, int n_in,
                              void* d_out, int out_size, void* d_ws, size_t ws_size,
                              hipStream_t stream) {
  (void)in_sizes; (void)n_in; (void)out_size; (void)ws_size;
  const float* q  = (const float*)d_in[0];
  const float* k  = (const float*)d_in[1];
  const float* v  = (const float*)d_in[2];
  const float* Wq = (const float*)d_in[3];
  const float* bq = (const float*)d_in[4];
  const float* Wk = (const float*)d_in[5];
  const float* bk = (const float*)d_in[6];
  const float* Wv = (const float*)d_in[7];
  const float* bv = (const float*)d_in[8];
  const float* Wo = (const float*)d_in[9];
  const float* bo = (const float*)d_in[10];

  char* ws = (char*)d_ws;
  const size_t SZ  = (size_t)B_ * S_ * D_ * sizeof(bf16);  // 8 MB
  const size_t WSZ = (size_t)D_ * D_ * sizeof(bf16);       // 2 MB
  bf16*  qh   = (bf16*)(ws);
  bf16*  kh   = (bf16*)(ws + SZ);
  bf16*  vhT  = (bf16*)(ws + 2 * SZ);
  bf16*  ctxb = (bf16*)(ws + 3 * SZ);
  bf16*  xb   = (bf16*)(ws + 4 * SZ);            // reused activation buffer
  bf16*  Wb   = (bf16*)(ws + 5 * SZ);            // reused weight buffer
  float* mcol = (float*)(ws + 5 * SZ + WSZ);
  float* zcol = (float*)(ws + 5 * SZ + WSZ + (size_t)BH_ * S_ * sizeof(float));

  const int NX8 = (B_ * S_ * D_) / 8;   // activation elems / 8
  const int NW8 = (D_ * D_) / 8;        // weight elems / 8
  const dim3 blk(32, 4);
  const dim3 gP(16, 64);                // proj / oproj grids
  const dim3 gA(32, 32);                // stats / ctx grids

  // Q path
  k_cvt <<<NX8 / 256, 256, 0, stream>>>(q, xb, NX8);
  k_cvt <<<NW8 / 256, 256, 0, stream>>>(Wq, Wb, NW8);
  k_proj<<<gP, blk, 0, stream>>>(xb, Wb, bq, qh, 0);
  // K path
  k_cvt <<<NX8 / 256, 256, 0, stream>>>(k, xb, NX8);
  k_cvt <<<NW8 / 256, 256, 0, stream>>>(Wk, Wb, NW8);
  k_proj<<<gP, blk, 0, stream>>>(xb, Wb, bk, kh, 0);
  // V path (stored transposed [b,h,hd,s])
  k_cvt <<<NX8 / 256, 256, 0, stream>>>(v, xb, NX8);
  k_cvt <<<NW8 / 256, 256, 0, stream>>>(Wv, Wb, NW8);
  k_proj<<<gP, blk, 0, stream>>>(xb, Wb, bv, vhT, 1);
  // attention
  k_stats<<<gA, blk, 0, stream>>>(qh, kh, mcol, zcol);
  k_ctx  <<<gA, blk, 0, stream>>>(qh, kh, vhT, mcol, zcol, ctxb);
  // output projection
  k_cvt  <<<NW8 / 256, 256, 0, stream>>>(Wo, Wb, NW8);
  k_oproj<<<gP, blk, 0, stream>>>(ctxb, Wb, bo, (float*)d_out);
}
